// LSTM_39608188404058
// MI455X (gfx1250) — compile-verified, matrix operand order checked by reading, not run on hardware
//
#include <hip/hip_runtime.h>

// ---------- types ----------
typedef __bf16 v16bf __attribute__((ext_vector_type(16)));
typedef float  v8f   __attribute__((ext_vector_type(8)));
typedef unsigned short u16x8 __attribute__((ext_vector_type(8)));

// ---------- problem constants ----------
#define P_    8192
#define L_    8
#define H_    660      // hidden
#define INW_  310      // raw LSTM input dim
#define KX_   320      // padded input dim (mult of 32)
#define KH_   672      // padded hidden dim (mult of 32, 42 groups of 16)
#define KC1_  992      // KX_+KH_  (layer-1 fused K)
#define KC2_  1344     // 2*KH_    (layer-2 fused K)
#define NP_   2688     // 4*KH_ = packed gate-interleaved N (21 * 128)
#define E_    300
#define B_    256
#define DTOT_ 1260     // 2*E + H

// ---------- helpers ----------
__device__ __forceinline__ unsigned short f2bf(float x) {
  unsigned int u = __float_as_uint(x);
  unsigned int r = (u + 0x7FFFu + ((u >> 16) & 1u)) >> 16;
  return (unsigned short)r;
}
__device__ __forceinline__ float sigm(float x) { return 1.0f / (1.0f + __expf(-x)); }

// ---------- zero init (d_ws is poisoned) ----------
__global__ void zero_kernel(unsigned int* __restrict__ p, size_t n) {
  size_t i = (size_t)blockIdx.x * blockDim.x + threadIdx.x;
  size_t stride = (size_t)gridDim.x * blockDim.x;
  for (; i < n; i += stride) p[i] = 0u;
}

// ---------- weight packing: gate-interleaved rows, bf16, zero-padded K ----------
// packed row r: group=r/64, gate=(r/16)%4, unit=group*16 + r%16
// layer1 K layout: [x(310) pad(10) | h(660) pad(12)]   K=992
__global__ void wpack1_kernel(const float* __restrict__ Wih1, const float* __restrict__ Whh1,
                              unsigned short* __restrict__ Wc) {
  int idx = blockIdx.x * blockDim.x + threadIdx.x;
  if (idx >= NP_ * KC1_) return;
  int r = idx / KC1_, k = idx - r * KC1_;
  int gt = (r >> 4) & 3, u = (r >> 6) * 16 + (r & 15);
  float v = 0.0f;
  if (u < H_) {
    int R = gt * H_ + u;
    if (k < INW_)                           v = Wih1[(size_t)R * INW_ + k];
    else if (k >= KX_ && (k - KX_) < H_)    v = Whh1[(size_t)R * H_ + (k - KX_)];
  }
  Wc[idx] = f2bf(v);
}
// layer2 K layout: [h1(660) pad(12) | h2(660) pad(12)]  K=1344
__global__ void wpack2_kernel(const float* __restrict__ Wih2, const float* __restrict__ Whh2,
                              unsigned short* __restrict__ Wc) {
  int idx = blockIdx.x * blockDim.x + threadIdx.x;
  if (idx >= NP_ * KC2_) return;
  int r = idx / KC2_, k = idx - r * KC2_;
  int gt = (r >> 4) & 3, u = (r >> 6) * 16 + (r & 15);
  float v = 0.0f;
  if (u < H_) {
    int R = gt * H_ + u;
    if (k < H_)                             v = Wih2[(size_t)R * H_ + k];
    else if (k >= KH_ && (k - KH_) < H_)    v = Whh2[(size_t)R * H_ + (k - KH_)];
  }
  Wc[idx] = f2bf(v);
}
__global__ void bpack_kernel(const float* __restrict__ b, float* __restrict__ bp) {
  int r = blockIdx.x * blockDim.x + threadIdx.x;
  if (r >= NP_) return;
  int gt = (r >> 4) & 3, u = (r >> 6) * 16 + (r & 15);
  bp[r] = (u < H_) ? b[gt * H_ + u] : 0.0f;
}

// ---------- embeddings: lookup + per-vector L2 norm -> bf16 X [P][L][KX_] ----------
// one wave (32 lanes) per (p,l)
__global__ __launch_bounds__(256) void embed_kernel(
    const int* __restrict__ widx, const int* __restrict__ pidx,
    const int* __restrict__ didx, const int* __restrict__ ridx,
    const float* __restrict__ wemb, const float* __restrict__ pemb,
    const float* __restrict__ demb, const float* __restrict__ remb,
    unsigned short* __restrict__ Xb) {
  int wid  = blockIdx.x * 8 + (threadIdx.x >> 5);
  int lane = threadIdx.x & 31;
  if (wid >= P_ * L_) return;
  const int wi = widx[wid], pi = pidx[wid], di = didx[wid], ri = ridx[wid];
  unsigned short* out = Xb + (size_t)wid * KX_;

  // word (300)
  const float* we = wemb + (size_t)wi * E_;
  float s = 0.0f;
  for (int j = lane; j < E_; j += 32) { float v = we[j]; s += v * v; }
  for (int off = 16; off; off >>= 1) s += __shfl_xor(s, off, 32);
  float inv = rsqrtf(s);
  for (int j = lane; j < E_; j += 32) out[j] = f2bf(we[j] * inv);

  // pos (4)
  float sp = 0.0f, pv = 0.0f;
  if (lane < 4) { pv = pemb[(size_t)pi * 4 + lane]; sp = pv * pv; }
  for (int off = 16; off; off >>= 1) sp += __shfl_xor(sp, off, 32);
  if (lane < 4) out[E_ + lane] = f2bf(pv * rsqrtf(sp));

  // dep (5)
  float sd = 0.0f, dv = 0.0f;
  if (lane < 5) { dv = demb[(size_t)di * 5 + lane]; sd = dv * dv; }
  for (int off = 16; off; off >>= 1) sd += __shfl_xor(sd, off, 32);
  if (lane < 5) out[E_ + 4 + lane] = f2bf(dv * rsqrtf(sd));

  // dir (1) + pad
  if (lane == 0) { float rv = remb[ri]; out[E_ + 9] = f2bf(rv * rsqrtf(rv * rv)); }
  if (lane < KX_ - INW_) out[INW_ + lane] = 0;
}

// ---------- fused WMMA GEMM + LSTM cell ----------
// D[M=8192, N=2688] = [A0|A1][M,K] * Bw[N,K]^T, gates interleaved per 64-col group.
// 8 waves/block, waves 4(M) x 2(N-group): block tile 128x128. Grid (21, 64), exact.
// Each wave: 32 rows x 1 group(64 cols) = 2x4 tiles of 16x16 -> 8 WMMA / 12 loads per K-step.
// K loop split into two phases (A0 then A1) so address chains are loop-invariant.
// Epilogue: per-lane element-wise LSTM cell update (i,f,g,o live in the 4 N-tiles).
__global__ __launch_bounds__(256) void lstm_gemm_fused(
    const unsigned short* __restrict__ A0, int lda0, int K0,
    const unsigned short* __restrict__ A1, int lda1, int K1,
    const unsigned short* __restrict__ Bw,   // [NP_][K0+K1]
    const float* __restrict__ biasP,
    float* __restrict__ cState,
    unsigned short* __restrict__ hOut,
    const int* __restrict__ lengths,
    float* __restrict__ last, int t) {
  const int tid  = threadIdx.x;
  const int lane = tid & 31;
  const int wave = tid >> 5;
  const int wn = wave & 1;
  const int wm = wave >> 1;
  const int m0   = blockIdx.y * 128 + wm * 32;
  const int gIdx = blockIdx.x * 2 + wn;        // 16-unit group, 0..41
  const int n0   = gIdx * 64;
  const int mr   = lane & 15;
  const int sub8 = ((lane >> 4) & 1) * 8;
  const int K    = K0 + K1;

  const unsigned short* aRow0[2];
  const unsigned short* aRow1[2];
  const unsigned short* bRow[4];
#pragma unroll
  for (int i = 0; i < 2; ++i) {
    int r = m0 + i * 16 + mr;
    aRow0[i] = A0 + (size_t)r * lda0 + sub8;
    aRow1[i] = A1 + (size_t)r * lda1 + sub8;
  }
#pragma unroll
  for (int j = 0; j < 4; ++j)
    bRow[j] = Bw + (size_t)(n0 + j * 16 + mr) * K + sub8;

  v8f acc[2][4] = {};

  // phase 0: A operand from A0, K range [0, K0)
  for (int kb = 0; kb < K0; kb += 32) {
    union { v16bf v; u16x8 h[2]; } af[2], bfr[4];
#pragma unroll
    for (int i = 0; i < 2; ++i) {
      af[i].h[0] = *(const u16x8*)(aRow0[i] + kb);
      af[i].h[1] = *(const u16x8*)(aRow0[i] + kb + 16);
    }
#pragma unroll
    for (int j = 0; j < 4; ++j) {
      bfr[j].h[0] = *(const u16x8*)(bRow[j] + kb);
      bfr[j].h[1] = *(const u16x8*)(bRow[j] + kb + 16);
    }
#pragma unroll
    for (int i = 0; i < 2; ++i)
#pragma unroll
      for (int j = 0; j < 4; ++j)
        acc[i][j] = __builtin_amdgcn_wmma_f32_16x16x32_bf16(
            false, af[i].v, false, bfr[j].v, (short)0, acc[i][j], false, false);
  }

  // phase 1: A operand from A1, K range [K0, K)
  for (int kb = 0; kb < K1; kb += 32) {
    union { v16bf v; u16x8 h[2]; } af[2], bfr[4];
#pragma unroll
    for (int i = 0; i < 2; ++i) {
      af[i].h[0] = *(const u16x8*)(aRow1[i] + kb);
      af[i].h[1] = *(const u16x8*)(aRow1[i] + kb + 16);
    }
#pragma unroll
    for (int j = 0; j < 4; ++j) {
      bfr[j].h[0] = *(const u16x8*)(bRow[j] + K0 + kb);
      bfr[j].h[1] = *(const u16x8*)(bRow[j] + K0 + kb + 16);
    }
#pragma unroll
    for (int i = 0; i < 2; ++i)
#pragma unroll
      for (int j = 0; j < 4; ++j)
        acc[i][j] = __builtin_amdgcn_wmma_f32_16x16x32_bf16(
            false, af[i].v, false, bfr[j].v, (short)0, acc[i][j], false, false);
  }

  // ---- fused LSTM cell epilogue (register-local: tiles j=0..3 are i,f,g,o) ----
  const int unit = gIdx * 16 + mr;
  const float bI = biasP[n0 + mr];
  const float bF = biasP[n0 + 16 + mr];
  const float bG = biasP[n0 + 32 + mr];
  const float bO = biasP[n0 + 48 + mr];
  const int rowOff = ((lane >> 4) & 1) * 8;
#pragma unroll
  for (int i = 0; i < 2; ++i) {
#pragma unroll
    for (int v = 0; v < 8; ++v) {
      int r = m0 + i * 16 + rowOff + v;
      size_t ci = (size_t)r * KH_ + unit;
      float iv = sigm (acc[i][0][v] + bI);
      float fv = sigm (acc[i][1][v] + bF);
      float gv = tanhf(acc[i][2][v] + bG);
      float ov = sigm (acc[i][3][v] + bO);
      float c  = fv * cState[ci] + iv * gv;
      cState[ci] = c;
      float h = ov * tanhf(c);
      hOut[ci] = f2bf(h);
      if (last != nullptr && unit < H_ && (lengths[r] - 1 == t))
        last[(size_t)r * H_ + unit] = h;
    }
  }
}

// ---------- segment sums ----------
__global__ void segsum_kernel(const float* __restrict__ last, const float* __restrict__ counts,
                              const int* __restrict__ path2ex, float* __restrict__ num) {
  int idx = blockIdx.x * blockDim.x + threadIdx.x;
  if (idx >= P_ * H_) return;
  int p = idx / H_, j = idx - p * H_;
  atomicAdd(&num[(size_t)path2ex[p] * H_ + j], last[idx] * counts[p]);
}
__global__ void segden_kernel(const float* __restrict__ counts, const int* __restrict__ path2ex,
                              float* __restrict__ den) {
  int p = blockIdx.x * blockDim.x + threadIdx.x;
  if (p >= P_) return;
  atomicAdd(&den[path2ex[p]], counts[p]);
}

// ---------- classifier + log_softmax: one wave per example ----------
__global__ __launch_bounds__(32) void out_kernel(
    const int* __restrict__ pairs, const float* __restrict__ wemb,
    const float* __restrict__ num, const float* __restrict__ den,
    const float* __restrict__ W_out, const float* __restrict__ b_out,
    float* __restrict__ out) {
  int b = blockIdx.x, lane = threadIdx.x;
  float inv = 1.0f / fmaxf(den[b], 1e-6f);
  int w0 = pairs[b * 2], w1 = pairs[b * 2 + 1];
  float a0 = 0, a1 = 0, a2 = 0, a3 = 0;
  for (int j = lane; j < DTOT_; j += 32) {
    float hv;
    if (j < E_)             hv = wemb[(size_t)w0 * E_ + j];
    else if (j < E_ + H_)   hv = num[(size_t)b * H_ + (j - E_)] * inv;
    else                    hv = wemb[(size_t)w1 * E_ + (j - E_ - H_)];
    a0 += hv * W_out[j];
    a1 += hv * W_out[DTOT_ + j];
    a2 += hv * W_out[2 * DTOT_ + j];
    a3 += hv * W_out[3 * DTOT_ + j];
  }
  for (int off = 16; off; off >>= 1) {
    a0 += __shfl_xor(a0, off, 32);
    a1 += __shfl_xor(a1, off, 32);
    a2 += __shfl_xor(a2, off, 32);
    a3 += __shfl_xor(a3, off, 32);
  }
  if (lane == 0) {
    float l0 = a0 + b_out[0], l1 = a1 + b_out[1], l2 = a2 + b_out[2], l3 = a3 + b_out[3];
    float m = fmaxf(fmaxf(l0, l1), fmaxf(l2, l3));
    float s = expf(l0 - m) + expf(l1 - m) + expf(l2 - m) + expf(l3 - m);
    float lse = m + logf(s);
    out[b * 4 + 0] = l0 - lse;
    out[b * 4 + 1] = l1 - lse;
    out[b * 4 + 2] = l2 - lse;
    out[b * 4 + 3] = l3 - lse;
  }
}

// ---------- workspace layout (bytes, all 256-aligned) ----------
static const size_t SZ_XB  = (size_t)P_ * L_ * KX_ * 2;   //  41,943,040
static const size_t SZ_H   = (size_t)P_ * KH_ * 2;        //  11,010,048 (x4: h1 a/b, h2 a/b)
static const size_t SZ_C   = (size_t)P_ * KH_ * 4;        //  22,020,096 (x2: c1, c2)
static const size_t SZ_LST = (size_t)P_ * H_ * 4;         //  21,626,880
static const size_t SZ_NUM = (size_t)B_ * H_ * 4;         //     675,840
static const size_t SZ_DEN = (size_t)B_ * 4;
static const size_t SZ_W1  = (size_t)NP_ * KC1_ * 2;      //   5,332,992
static const size_t SZ_W2  = (size_t)NP_ * KC2_ * 2;      //   7,225,344
static const size_t SZ_BP  = (size_t)NP_ * 4;             //      10,752

extern "C" void kernel_launch(void* const* d_in, const int* in_sizes, int n_in,
                              void* d_out, int out_size, void* d_ws, size_t ws_size,
                              hipStream_t stream) {
  const int*   word_idx = (const int*)d_in[0];
  const int*   pos_idx  = (const int*)d_in[1];
  const int*   dep_idx  = (const int*)d_in[2];
  const int*   dir_idx  = (const int*)d_in[3];
  const int*   lengths  = (const int*)d_in[4];
  const int*   path2ex  = (const int*)d_in[5];
  const int*   pairs    = (const int*)d_in[6];
  const float* counts   = (const float*)d_in[7];
  const float* word_emb = (const float*)d_in[8];
  const float* pos_emb  = (const float*)d_in[9];
  const float* dep_emb  = (const float*)d_in[10];
  const float* dir_emb  = (const float*)d_in[11];
  const float* W_ih1    = (const float*)d_in[12];
  const float* W_hh1    = (const float*)d_in[13];
  const float* b1       = (const float*)d_in[14];
  const float* W_ih2    = (const float*)d_in[15];
  const float* W_hh2    = (const float*)d_in[16];
  const float* b2       = (const float*)d_in[17];
  const float* W_out    = (const float*)d_in[18];
  const float* b_out    = (const float*)d_in[19];
  float* out = (float*)d_out;

  char* ws = (char*)d_ws;
  size_t off = 0;
  unsigned short* Xb  = (unsigned short*)(ws + off); off += SZ_XB;
  unsigned short* h1a = (unsigned short*)(ws + off); off += SZ_H;
  unsigned short* h1b = (unsigned short*)(ws + off); off += SZ_H;
  unsigned short* h2a = (unsigned short*)(ws + off); off += SZ_H;
  unsigned short* h2b = (unsigned short*)(ws + off); off += SZ_H;
  float*          c1  = (float*)(ws + off);          off += SZ_C;
  float*          c2  = (float*)(ws + off);          off += SZ_C;
  float*          last= (float*)(ws + off);          off += SZ_LST;
  float*          num = (float*)(ws + off);          off += SZ_NUM;
  float*          den = (float*)(ws + off);          off += SZ_DEN;
  unsigned short* Wc1 = (unsigned short*)(ws + off); off += SZ_W1;
  unsigned short* Wc2 = (unsigned short*)(ws + off); off += SZ_W2;
  float*          bp1 = (float*)(ws + off);          off += SZ_BP;
  float*          bp2 = (float*)(ws + off);          off += SZ_BP;
  (void)ws_size; (void)n_in; (void)in_sizes; (void)out_size;

  // zero: h1a..h2b (contiguous), c1+c2 (contiguous), num+den (contiguous)
  zero_kernel<<<4096, 256, 0, stream>>>((unsigned int*)h1a, (4 * SZ_H) / 4);
  zero_kernel<<<4096, 256, 0, stream>>>((unsigned int*)c1,  (2 * SZ_C) / 4);
  zero_kernel<<<64,   256, 0, stream>>>((unsigned int*)num, (SZ_NUM + SZ_DEN) / 4);

  // weight/bias packing (bf16, gate-interleaved, padded K)
  wpack1_kernel<<<(NP_ * KC1_ + 255) / 256, 256, 0, stream>>>(W_ih1, W_hh1, Wc1);
  wpack2_kernel<<<(NP_ * KC2_ + 255) / 256, 256, 0, stream>>>(W_ih2, W_hh2, Wc2);
  bpack_kernel<<<(NP_ + 255) / 256, 256, 0, stream>>>(b1, bp1);
  bpack_kernel<<<(NP_ + 255) / 256, 256, 0, stream>>>(b2, bp2);

  // embeddings -> normalized bf16 inputs
  embed_kernel<<<(P_ * L_) / 8, 256, 0, stream>>>(word_idx, pos_idx, dep_idx, dir_idx,
                                                  word_emb, pos_emb, dep_emb, dir_emb, Xb);

  unsigned short* h1[2] = {h1a, h1b};
  unsigned short* h2[2] = {h2a, h2b};
  dim3 gemmGrid(NP_ / 128, P_ / 128);   // (21, 64), exact

  for (int t = 0; t < L_; ++t) {
    int cur = t & 1, nxt = cur ^ 1;
    // layer 1: gates = [x_t | h1_cur] * Wc1^T  -> c1, h1_nxt
    lstm_gemm_fused<<<gemmGrid, 256, 0, stream>>>(
        Xb + (size_t)t * KX_, L_ * KX_, KX_,
        h1[cur], KH_, KH_,
        Wc1, bp1, c1, h1[nxt],
        nullptr, nullptr, t);
    // layer 2: gates = [h1_nxt | h2_cur] * Wc2^T -> c2, h2_nxt (+ last capture)
    lstm_gemm_fused<<<gemmGrid, 256, 0, stream>>>(
        h1[nxt], KH_, KH_,
        h2[cur], KH_, KH_,
        Wc2, bp2, c2, h2[nxt],
        lengths, last, t);
  }

  // segment aggregation
  segsum_kernel<<<(P_ * H_ + 255) / 256, 256, 0, stream>>>(last, counts, path2ex, num);
  segden_kernel<<<(P_ + 255) / 256, 256, 0, stream>>>(counts, path2ex, den);

  // classifier + log_softmax
  out_kernel<<<B_, 32, 0, stream>>>(pairs, word_emb, num, den, W_out, b_out, out);
}
